// CELoss_4896262717859
// MI455X (gfx1250) — matrix operand order repaired and checked by reading, not compiled, not run on hardware
//
#include <hip/hip_runtime.h>
#include <hip/hip_bf16.h>
#include <math.h>

typedef __attribute__((ext_vector_type(16))) _Float16 v16h;
typedef __attribute__((ext_vector_type(8)))  _Float16 v8h;
typedef __attribute__((ext_vector_type(8)))  float    v8f;

#define Nn 8192
#define Kk 4096
#define CBLOCKS (Kk / 16)          // 256 column blocks of 16 idx-columns
#define SLABS   64                 // row slabs per column block
#define ROWS_PER_SLAB (Nn / SLABS) // 128 rows -> 4 chunks of 32

// Workspace layout (bytes):
//   acc : float[8*Kk]     @ 0       (131072 B)  acc[(side*4+stat)*Kk + c]
//   bpl : _Float16[5*Nn]  @ 131072  ( 81920 B)  planes: 0 exp(out), 1 pos?out:0,
//                                               2 one, 3 pos?1:0, 4 zeros
#define ACC_FLOATS (8 * Kk)
#define BPL_OFFSET (ACC_FLOATS * sizeof(float))

__global__ void ce_init_kernel(const float* __restrict__ outputs,
                               const int*   __restrict__ targets,
                               float* __restrict__ acc,
                               _Float16* __restrict__ bpl,
                               float* __restrict__ d_out) {
    int t = blockIdx.x * blockDim.x + threadIdx.x;
    if (t < ACC_FLOATS) acc[t] = 0.0f;
    if (t < Nn) {
        float o = outputs[t];
        bool  p = targets[t] != 0;
        bpl[0 * Nn + t] = (_Float16)expf(o);           // for logsumexp
        bpl[1 * Nn + t] = (_Float16)(p ? o : 0.0f);    // pos_logit
        bpl[2 * Nn + t] = (_Float16)1.0f;              // cnt
        bpl[3 * Nn + t] = (_Float16)(p ? 1.0f : 0.0f); // pos_cnt
        bpl[4 * Nn + t] = (_Float16)0.0f;              // zero plane (lanes n>=4)
    }
    if (t == 0) *d_out = 0.0f;
}

// One wave per block. Each wave owns 16 idx-columns (cBase..cBase+15) and a
// 128-row slab; per 32-row chunk it gathers the adjacency tile, builds the
// lower/upper {0,1} f16 A-matrices and the statistics B-matrix, and feeds two
// v_wmma_f32_16x16x32_f16 accumulating 4 statistics x 16 columns per side.
__global__ void __launch_bounds__(32)
ce_wmma_kernel(const int* __restrict__ node_adj,
               const int* __restrict__ idx_node,
               const _Float16* __restrict__ bpl,
               float* __restrict__ acc) {
    const int  lane  = threadIdx.x;        // 0..31
    const int  m     = lane & 15;          // A row / D column, B column
    const int  n     = m;                  // statistic index for this lane
    const bool lo    = lane < 16;
    const int  cBase = blockIdx.x * 16;
    const int  j     = idx_node[cBase + m];        // this A-row's adjacency column
    const int* pj    = node_adj + j;

    // K-offsets, contiguous k = 16*(lane/16) + s in both A and B (wave32):
    const int kA0 = lo ? 0  : 16;  // A halves 0..7
    const int kA1 = lo ? 8  : 24;  // A halves 8..15
    const int kB  = lo ? 0  : 16;  // B halves 0..15

    // Lanes with n>=4 read the zero plane: unconditional, divergence-free loads.
    const v8h* bq = (const v8h*)(bpl + (size_t)(n < 4 ? n : 4) * Nn + kB);

    v8f dLow = {};
    v8f dUp  = {};

    const int rowBeg = blockIdx.y * ROWS_PER_SLAB;
    const int rowEnd = rowBeg + ROWS_PER_SLAB;

    for (int ib = rowBeg; ib < rowEnd; ib += 32) {
        // B: 16 contiguous halves per lane -> two global_load_b128
        v8h bA = bq[ib / 8];
        v8h bB = bq[ib / 8 + 1];
        v16h b = __builtin_shufflevector(bA, bB, 0, 1, 2, 3, 4, 5, 6, 7,
                                                 8, 9, 10, 11, 12, 13, 14, 15);
        v16h aL, aU;
        const _Float16 h1 = (_Float16)1.0f;
        const _Float16 h0 = (_Float16)0.0f;
#pragma unroll
        for (int h = 0; h < 8; ++h) {
            int i0 = ib + kA0 + h;
            int i1 = ib + kA1 + h;
            int v0 = pj[(size_t)i0 * Nn];   // immediate-offset strided gathers
            int v1 = pj[(size_t)i1 * Nn];
            aL[h]     = (v0 != 0 && i0 < j) ? h1 : h0;
            aU[h]     = (v0 != 0 && i0 > j) ? h1 : h0;
            aL[8 + h] = (v1 != 0 && i1 < j) ? h1 : h0;
            aU[8 + h] = (v1 != 0 && i1 > j) ? h1 : h0;
        }
        // D[m][n] += sum_k A[m][k] * B[k][n]  (f32 accumulate)
        dLow = __builtin_amdgcn_wmma_f32_16x16x32_f16(false, aL, false, b,
                                                      (short)0, dLow, false, false);
        dUp  = __builtin_amdgcn_wmma_f32_16x16x32_f16(false, aU, false, b,
                                                      (short)0, dUp,  false, false);
    }

    // D layout: VGPR r -> column (lo ? r : 8+r), statistic n = lane&15.
    if (n < 4) {
#pragma unroll
        for (int r = 0; r < 8; ++r) {
            int mm = lo ? r : (8 + r);
            atomicAdd(&acc[(size_t)(0 * 4 + n) * Kk + cBase + mm], dLow[r]);
            atomicAdd(&acc[(size_t)(1 * 4 + n) * Kk + cBase + mm], dUp[r]);
        }
    }
}

__global__ void ce_finalize_kernel(const float* __restrict__ acc,
                                   float* __restrict__ d_out) {
    __shared__ float red[256];
    int t    = blockIdx.x * blockDim.x + threadIdx.x;  // 0 .. 2*Kk-1
    int side = t >> 12;                                // t / Kk
    int c    = t & (Kk - 1);
    const float* a = acc + (size_t)side * 4 * Kk;
    float sumexp = a[0 * Kk + c];
    float poslog = a[1 * Kk + c];
    float cnt    = a[2 * Kk + c];
    float pcnt   = a[3 * Kk + c];
    float contrib = 0.0f;
    // counts are exact sums of 1.0f -> integer-valued floats
    if (cnt > 0.5f && pcnt > 0.5f && pcnt < 1.5f) {
        contrib = (logf(sumexp) - poslog) / cnt;
    }
    red[threadIdx.x] = contrib;
    __syncthreads();
    for (int s = 128; s > 0; s >>= 1) {
        if (threadIdx.x < s) red[threadIdx.x] += red[threadIdx.x + s];
        __syncthreads();
    }
    if (threadIdx.x == 0) atomicAdd(d_out, red[0]);
}

extern "C" void kernel_launch(void* const* d_in, const int* in_sizes, int n_in,
                              void* d_out, int out_size, void* d_ws, size_t ws_size,
                              hipStream_t stream) {
    (void)in_sizes; (void)n_in; (void)out_size; (void)ws_size;
    const float* outputs  = (const float*)d_in[0];
    const int*   targets  = (const int*)  d_in[1];
    const int*   node_adj = (const int*)  d_in[2];
    const int*   idx_node = (const int*)  d_in[3];
    float*       out      = (float*)d_out;

    float*    acc = (float*)d_ws;
    _Float16* bpl = (_Float16*)((char*)d_ws + BPL_OFFSET);

    // Zero accumulators + build f16 statistic planes + zero output (ws/out are poisoned).
    ce_init_kernel<<<(ACC_FLOATS + 255) / 256, 256, 0, stream>>>(outputs, targets, acc, bpl, out);

    dim3 grid(CBLOCKS, SLABS);
    ce_wmma_kernel<<<grid, 32, 0, stream>>>(node_adj, idx_node, bpl, acc);

    ce_finalize_kernel<<<(2 * Kk) / 256, 256, 0, stream>>>(acc, out);
}
